// SpatialGraphConv_14886356648730
// MI455X (gfx1250) — matrix operand order repaired and checked by reading, not compiled, not run on hardware
//
#include <hip/hip_runtime.h>

// Problem constants (fixed by the reference)
#define KSZ    3
#define COUT   128
#define CIN    64
#define NB     64
#define TDIM   256
#define VDIM   25
#define TT     4                 // t-values per block
#define JDIM   (KSZ * CIN)       // 192 : fused (k,ci) K-dim of stage B

// LDS layout (bytes) — first 71680 B mirror the ws blob written by prep
#define OFF_W2    0              // 128*192 bf16  = 49152 B
#define OFF_EFFA  49152          // 3*32*32 bf16  =  6144 B (effA[k][w][v], padded, transposed)
#define OFF_BEFF  55296          // 128*32  f32   = 16384 B
#define OFF_XS    71680          // TT*64*32 bf16 = 16384 B (x[t'][ci][v pad32])
#define OFF_XA    88064          // TT*32*192 bf16= 49152 B (xa[t'][w pad32][k*64+ci])
#define BLOB_B    71680          // bytes copied async from ws
#define BLOB_Q    (BLOB_B / 16)  // 4480 b128 transfers
#define SMEM_BYTES 137216

typedef __attribute__((ext_vector_type(16))) __bf16 v16bf;
typedef __attribute__((ext_vector_type(8)))  float  v8f;

struct U4 { unsigned int x, y, z, w; };
union FragBF { v16bf v; U4 q[2]; };

__device__ __forceinline__ unsigned short f2bf(float f) {
    unsigned int u = __float_as_uint(f);
    unsigned int r = u + 0x7FFFu + ((u >> 16) & 1u);   // round-to-nearest-even
    return (unsigned short)(r >> 16);
}

// ---------------------------------------------------------------------------
// Prep: fold W -> W2 bf16 (K-dim j = k*64+ci), effA -> transposed padded bf16,
// and bias -> beff[c][w] = sum_k b[k*128+c] * sum_v effA[k,v,w]  (f32)
// ---------------------------------------------------------------------------
__global__ void sgc_prep(const float* __restrict__ W, const float* __restrict__ b,
                         const float* __restrict__ A, const float* __restrict__ edge,
                         const float* __restrict__ adA,
                         unsigned short* __restrict__ w2,
                         unsigned short* __restrict__ effAs,
                         float* __restrict__ beff) {
    int gid = blockIdx.x * blockDim.x + threadIdx.x;
    int gsz = gridDim.x * blockDim.x;

    for (int i = gid; i < COUT * JDIM; i += gsz) {           // W2[c][j]
        int c = i / JDIM, j = i % JDIM;
        int k = j >> 6, ci = j & 63;
        w2[i] = f2bf(W[(k * COUT + c) * CIN + ci]);
    }
    for (int i = gid; i < KSZ * 32 * 32; i += gsz) {         // effAs[k][w][v]
        int k = i / 1024, w = (i >> 5) & 31, v = i & 31;
        float val = 0.f;
        if (v < VDIM && w < VDIM) {
            int idx = (k * VDIM + v) * VDIM + w;
            val = A[idx] * edge[idx] + adA[idx];
        }
        effAs[i] = f2bf(val);
    }
    for (int i = gid; i < COUT * 32; i += gsz) {             // beff[c][w]
        int c = i >> 5, w = i & 31;
        float acc = 0.f;
        if (w < VDIM) {
            for (int k = 0; k < KSZ; ++k) {
                float s = 0.f;
                for (int v = 0; v < VDIM; ++v) {
                    int idx = (k * VDIM + v) * VDIM + w;
                    s += A[idx] * edge[idx] + adA[idx];
                }
                acc += b[k * COUT + c] * s;
            }
        }
        beff[i] = acc;
    }
}

// ---------------------------------------------------------------------------
// Fused main kernel: block = one (n, 4-t tile); 256 threads = 8 wave32 waves.
// Stage A: xa[t'][w][k*64+ci] = sum_v x[ci,t',v] * effA[k,v,w]   (WMMA bf16)
// Stage B: out[c,t',w] = beff[c,w] + sum_j W2[c,j] * xa[t'][w][j] (WMMA bf16)
// ---------------------------------------------------------------------------
__global__ void __launch_bounds__(256)
sgc_main(const float* __restrict__ x,
         const unsigned short* __restrict__ wsblob,   // W2 | effAs | beff, 71680 B
         float* __restrict__ out) {
    extern __shared__ __align__(16) char smem[];
    const int tid  = threadIdx.x;
    const int lane = tid & 31, wave = tid >> 5;
    const int hi   = lane >> 4, ln = lane & 15;
    const int tb   = blockIdx.x, n = blockIdx.y;
    const int t0   = tb * TT;

    // ---- async-copy precomputed tables ws -> LDS (ASYNCcnt path) ----
    {
        // generic LDS pointer: low 32 bits are the wave-relative LDS byte offset
        unsigned lds_base = (unsigned)(unsigned long long)(void*)smem;
        unsigned long long gbase = (unsigned long long)wsblob;
        #pragma unroll
        for (int it = 0; it < 18; ++it) {
            int idx = it * 256 + tid;              // tail is wave-uniform (waves 0-3)
            if (idx < BLOB_Q) {
                unsigned off = (unsigned)idx * 16u;
                unsigned laddr = lds_base + off;
                asm volatile("global_load_async_to_lds_b128 %0, %1, %2"
                             :: "v"(laddr), "v"(off), "s"(gbase)
                             : "memory");
            }
        }
    }
    // ---- fill LDS: x tile -> Xs[t'][ci][v pad32] bf16 (needs f32->bf16) ----
    {
        unsigned short* Xs = (unsigned short*)(smem + OFF_XS);
        for (int i = tid; i < TT * CIN * 32; i += 256) {
            int tp = i >> 11;
            int ci = (i >> 5) & 63;
            int v  = i & 31;
            float val = 0.f;
            if (v < VDIM)
                val = x[((n * CIN + ci) * TDIM + (t0 + tp)) * VDIM + v];
            Xs[i] = f2bf(val);
        }
    }
    asm volatile("s_wait_asynccnt 0x0" ::: "memory");
    __syncthreads();

    // ---- Stage A: 96 WMMA tiles (tp x k x ci-tile x w-tile), 12 per wave ----
    {
        const unsigned short* Xs = (const unsigned short*)(smem + OFF_XS);
        const unsigned short* Ea = (const unsigned short*)(smem + OFF_EFFA);
        unsigned int* Xa32 = (unsigned int*)(smem + OFF_XA);
        #pragma unroll
        for (int i = 0; i < 12; ++i) {
            const int tile = wave + i * 8;
            int wt = tile & 1;
            int ct = (tile >> 1) & 3;
            int k  = (tile >> 3) % KSZ;
            int tp = tile / 24;
            FragBF a, bm;
            {   // A frag: rows ci, K = v  (lanes 0-15: K 0-15; 16-31: K 16-31)
                int base = (tp * CIN + ct * 16 + ln) * 32;
                a.q[0] = *(const U4*)(Xs + base + hi * 16);
                a.q[1] = *(const U4*)(Xs + base + 8 + hi * 16);
            }
            int w = wt * 16 + ln;
            {   // B frag: cols w, K = v (lanes 0-15: K 0-15; 16-31: K 16-31)
                int base = (k * 32 + w) * 32 + hi * 16;
                bm.q[0] = *(const U4*)(Ea + base);
                bm.q[1] = *(const U4*)(Ea + base + 8);
            }
            v8f c = {};
            c = __builtin_amdgcn_wmma_f32_16x16x32_bf16(false, a.v, false, bm.v,
                                                        (short)0, c, false, false);
            // D rows are pairs of consecutive ci -> pack two bf16 per b32 store
            int eb = ((tp * 32 + w) * JDIM + k * CIN + ct * 16 + hi * 8) >> 1;
            #pragma unroll
            for (int q = 0; q < 4; ++q) {
                unsigned lo = f2bf(c[2 * q]);
                unsigned h  = f2bf(c[2 * q + 1]);
                Xa32[eb + q] = lo | (h << 16);
            }
        }
    }
    __syncthreads();

    // ---- Stage B: 64 tiles x 6 K-steps (K = 192 over (k,ci)), 8 per wave ----
    {
        const unsigned short* W2 = (const unsigned short*)(smem + OFF_W2);
        const unsigned short* Xa = (const unsigned short*)(smem + OFF_XA);
        const float*          Bf = (const float*)(smem + OFF_BEFF);
        #pragma unroll
        for (int i = 0; i < 8; ++i) {
            const int tile = wave + i * 8;
            int wt = tile & 1;
            int ct = (tile >> 1) & 7;
            int tp = tile >> 4;
            int w  = wt * 16 + ln;
            v8f c;
            #pragma unroll
            for (int r = 0; r < 8; ++r)          // seed accumulator with bias term
                c[r] = Bf[(ct * 16 + r + hi * 8) * 32 + w];
            #pragma unroll
            for (int ks = 0; ks < 6; ++ks) {
                FragBF a, bm;
                {
                    int base = (ct * 16 + ln) * JDIM + ks * 32;
                    a.q[0] = *(const U4*)(W2 + base + hi * 16);
                    a.q[1] = *(const U4*)(W2 + base + 8 + hi * 16);
                }
                {
                    int base = (tp * 32 + w) * JDIM + ks * 32 + hi * 16;
                    bm.q[0] = *(const U4*)(Xa + base);
                    bm.q[1] = *(const U4*)(Xa + base + 8);
                }
                c = __builtin_amdgcn_wmma_f32_16x16x32_bf16(false, a.v, false, bm.v,
                                                            (short)0, c, false, false);
            }
            if (w < VDIM) {
                #pragma unroll
                for (int r = 0; r < 8; ++r) {
                    int cc = ct * 16 + r + hi * 8;
                    out[((n * COUT + cc) * TDIM + (t0 + tp)) * VDIM + w] = c[r];
                }
            }
        }
    }
}

// ---------------------------------------------------------------------------
extern "C" void kernel_launch(void* const* d_in, const int* in_sizes, int n_in,
                              void* d_out, int out_size, void* d_ws, size_t ws_size,
                              hipStream_t stream) {
    const float* x    = (const float*)d_in[0];
    const float* W    = (const float*)d_in[1];
    const float* b    = (const float*)d_in[2];
    const float* A    = (const float*)d_in[3];
    const float* edge = (const float*)d_in[4];
    const float* adA  = (const float*)d_in[5];
    float* out = (float*)d_out;

    unsigned short* w2    = (unsigned short*)d_ws;                       // 49152 B
    unsigned short* effAs = (unsigned short*)((char*)d_ws + OFF_EFFA);   //  6144 B
    float*          beff  = (float*)((char*)d_ws + OFF_BEFF);            // 16384 B

    sgc_prep<<<dim3(32), dim3(256), 0, stream>>>(W, b, A, edge, adA, w2, effAs, beff);

    (void)hipFuncSetAttribute((const void*)sgc_main,
                              hipFuncAttributeMaxDynamicSharedMemorySize, SMEM_BYTES);
    sgc_main<<<dim3(TDIM / TT, NB), dim3(256), SMEM_BYTES, stream>>>(
        x, (const unsigned short*)d_ws, out);
}
